// CatNet_27419071217972
// MI455X (gfx1250) — matrix operand order repaired and checked by reading, not compiled, not run on hardware
//
#include <hip/hip_runtime.h>

// CatNet SNN forward for MI455X (gfx1250, wave32, WMMA).
// Spikes are binary -> exact in f16; weights f16, accumulate f32 via
// v_wmma_f32_16x16x32_f16. Membranes live in VGPRs across the sequential
// t-loop. Everything (<130MB) is L2-resident on MI455X (192MB L2).

typedef _Float16 h16;
typedef __attribute__((ext_vector_type(16))) _Float16 v16h;
typedef __attribute__((ext_vector_type(8)))  float    v8f;

#define TSTEPS 48
#define NB     32

// ---------------------------------------------------------------------------
// Weight prepack: f32 -> f16 B-fragments in exact CDNA5 WMMA lane order.
// Dst layout: [chunk][jtile][lane][16 halves]  (32B per lane, coalesced loads)
// B frag (16-bit 32x16): element idx -> K = (lane>=16?8:0)+idx+(idx>=8?8:0),
// N = jtile*16 + (lane&15).
// Conv K packing: k = kk*CIN + c (kk = ky*5+kx). cin==0 -> direct [O][K].
// ---------------------------------------------------------------------------
__global__ void prepack_kernel(const float* __restrict__ W, h16* __restrict__ dst,
                               int chunks, int jmax, int cin, int kreal) {
  int tid = blockIdx.x * blockDim.x + threadIdx.x;
  int total = chunks * jmax * 512;
  if (tid >= total) return;
  int idx  = tid & 15;
  int lane = (tid >> 4) & 31;
  int rest = tid >> 9;
  int j  = rest % jmax;
  int kc = rest / jmax;
  int o = j * 16 + (lane & 15);
  int k = kc * 32 + ((lane >= 16) ? 8 : 0) + idx + ((idx >= 8) ? 8 : 0);
  float v = 0.f;
  if (k < kreal) {
    if (cin > 0) {
      int c  = k % cin;
      int kk = k / cin;
      v = W[(o * cin + c) * 25 + kk];
    } else {
      v = W[o * kreal + k];
    }
  }
  dst[tid] = (h16)v;
}

// ---------------------------------------------------------------------------
// Layer 1: conv(1->16, 5x5, valid) + spike.  K=25 -> VALU, image in LDS.
// x: [N,1,32,32,T] (T innermost).  s1: [N][T][16][28][28] f16.
// grid = N*16 (one block per (n, oc)), 256 threads, 4 pixels/thread.
// ---------------------------------------------------------------------------
__global__ void conv1_spike_kernel(const float* __restrict__ x,
                                   const float* __restrict__ Wc1,
                                   const float* __restrict__ bc1,
                                   h16* __restrict__ s1) {
  const int n   = blockIdx.x >> 4;
  const int oc  = blockIdx.x & 15;
  const int tid = threadIdx.x;
  __shared__ float img[1024];
  float w[25];
#pragma unroll
  for (int i = 0; i < 25; ++i) w[i] = Wc1[oc * 25 + i];
  const float bias = bc1[oc];
  float u[4] = {0.f, 0.f, 0.f, 0.f};
  for (int t = 0; t < TSTEPS; ++t) {
    for (int e = tid; e < 1024; e += 256)
      img[e] = x[(n * 1024 + e) * TSTEPS + t];   // strided by T; L2-resident
    __syncthreads();
#pragma unroll
    for (int r = 0; r < 4; ++r) {
      int p = r * 256 + tid;
      if (p < 784) {
        int h = p / 28, wq = p - h * 28;
        float acc = bias;
#pragma unroll
        for (int ky = 0; ky < 5; ++ky)
#pragma unroll
          for (int kx = 0; kx < 5; ++kx)
            acc += w[ky * 5 + kx] * img[(h + ky) * 32 + (wq + kx)];
        u[r] += acc;
        float s = (u[r] >= 1.f) ? 1.f : 0.f;
        u[r] -= s;
        s1[((size_t)(n * TSTEPS + t) * 16 + oc) * 784 + p] = (h16)s;
      }
    }
    __syncthreads();
  }
}

// ---------------------------------------------------------------------------
// WMMA conv + spike (conv2 and conv3).  Each wave: one 4x4 output patch,
// all 32 output channels (2 N-tiles), sequential t-loop, membranes in VGPRs.
// A frag gathered from LDS-staged 8x8xCIN field; B frags preloaded packed.
// K packing k = kk*CIN + c -> c = k & (CIN-1), kk = k >> log2(CIN).
// ---------------------------------------------------------------------------
template<int CIN, int INH, int INW, int OUTH, int OUTW, int CHUNKS>
__global__ void conv_spike_wmma_kernel(const h16* __restrict__ sin,
                                       const h16* __restrict__ Bpack,
                                       const float* __restrict__ bias,
                                       h16* __restrict__ sout) {
  const int n    = blockIdx.y;
  const int wave = threadIdx.x >> 5;
  const int lane = threadIdx.x & 31;
  const int patch = blockIdx.x * 4 + wave;        // grids sized so always valid
  const int PW  = OUTW / 4;
  const int ph0 = (patch / PW) * 4;
  const int pw0 = (patch % PW) * 4;
  const int hi  = (lane >= 16) ? 1 : 0;
  const int m16 = lane & 15;                      // A-frag M row for this lane
  const int mh  = m16 >> 2, mw = m16 & 3;

  __shared__ h16 fld[4][CIN * 64];                // per-wave 8x8xCIN field
  h16* f = fld[wave];

  const float b0s = bias[m16];
  const float b1s = bias[16 + m16];

  v8f u0, u1;                                     // membranes (persist over t)
#pragma unroll
  for (int i = 0; i < 8; ++i) { u0[i] = 0.f; u1[i] = 0.f; }

  for (int t = 0; t < TSTEPS; ++t) {
    const h16* src = sin + (size_t)(n * TSTEPS + t) * CIN * INH * INW;
    for (int e = lane; e < CIN * 64; e += 32) {
      int c = e >> 6, yx = e & 63;
      int y = yx >> 3, xx = yx & 7;
      f[e] = src[c * (INH * INW) + (ph0 + y) * INW + (pw0 + xx)];
    }
    if (t + 1 < TSTEPS)                           // gfx1250 global_prefetch_b8
      __builtin_prefetch(sin + (size_t)(n * TSTEPS + t + 1) * CIN * INH * INW
                             + lane * 64, 0, 0);
    __syncthreads();

    v8f c0, c1;                                   // y = conv + bias
#pragma unroll
    for (int i = 0; i < 8; ++i) { c0[i] = b0s; c1[i] = b1s; }

    for (int kc = 0; kc < CHUNKS; ++kc) {
      v16h a;
#pragma unroll
      for (int idx = 0; idx < 16; ++idx) {
        // CDNA5 16-bit A 16x32 layout: K = (lane>=16?8:0)+idx+(idx>=8?8:0)
        int k = kc * 32 + hi * 8 + idx + ((idx >= 8) ? 8 : 0);
        bool valid = (k < 25 * CIN);              // pad tail chunk with zeros
        int c  = k % CIN;
        int kk = k / CIN;
        if (kk > 24) kk = 24;                     // clamp address, select 0 below
        int ky = kk / 5, kx = kk - ky * 5;
        h16 av = f[c * 64 + (mh + ky) * 8 + (mw + kx)];
        a[idx] = valid ? av : (h16)0.f;
      }
      const v16h* bp0 = (const v16h*)(Bpack + ((size_t)(kc * 2 + 0) * 32 + lane) * 16);
      const v16h* bp1 = (const v16h*)(Bpack + ((size_t)(kc * 2 + 1) * 32 + lane) * 16);
      c0 = __builtin_amdgcn_wmma_f32_16x16x32_f16(false, a, false, bp0[0],
                                                  (short)0, c0, false, false);
      c1 = __builtin_amdgcn_wmma_f32_16x16x32_f16(false, a, false, bp1[0],
                                                  (short)0, c1, false, false);
    }

    // membrane update + spike (reset by subtraction), store f16 spikes
    h16* dst = sout + (size_t)(n * TSTEPS + t) * 32 * OUTH * OUTW;
#pragma unroll
    for (int i = 0; i < 8; ++i) {
      int m = i + hi * 8;                         // C/D layout: M = vgpr + 8*hi
      int h = ph0 + (m >> 2), w = pw0 + (m & 3);
      u0[i] += c0[i];
      float sa = (u0[i] >= 1.f) ? 1.f : 0.f;
      u0[i] -= sa;
      dst[(size_t)m16 * (OUTH * OUTW) + h * OUTW + w] = (h16)sa;
      u1[i] += c1[i];
      float sb = (u1[i] >= 1.f) ? 1.f : 0.f;
      u1[i] -= sb;
      dst[(size_t)(16 + m16) * (OUTH * OUTW) + h * OUTW + w] = (h16)sb;
    }
    __syncthreads();                              // before next-t LDS overwrite
  }
}

// ---------------------------------------------------------------------------
// 2x2 avg-pool + spike. One thread per output neuron, t-loop, membrane in reg.
// ---------------------------------------------------------------------------
template<int INH, int INW>
__global__ void pool_spike_kernel(const h16* __restrict__ sin, h16* __restrict__ sout) {
  const int OH = INH / 2, OW = INW / 2;
  int id = blockIdx.x * blockDim.x + threadIdx.x;
  if (id >= NB * 32 * OH * OW) return;
  int w = id % OW;
  int h = (id / OW) % OH;
  int c = (id / (OW * OH)) % 32;
  int n = id / (OW * OH * 32);
  float u = 0.f;
  for (int t = 0; t < TSTEPS; ++t) {
    const h16* src = sin + ((size_t)(n * TSTEPS + t) * 32 + c) * (INH * INW);
    float y = 0.25f * ((float)src[(2 * h) * INW + 2 * w]
                     + (float)src[(2 * h) * INW + 2 * w + 1]
                     + (float)src[(2 * h + 1) * INW + 2 * w]
                     + (float)src[(2 * h + 1) * INW + 2 * w + 1]);
    u += y;
    float s = (u >= 1.f) ? 1.f : 0.f;
    u -= s;
    sout[((size_t)(n * TSTEPS + t) * 32 + c) * (OH * OW) + h * OW + w] = (h16)s;
  }
}

// ---------------------------------------------------------------------------
// fc1 (512->128) + spike as WMMA: M = batch (2 tiles of 16), N = 128 (8 tiles),
// K = 512 (16 chunks). One wave per (Mtile,Ntile) pair; grid = 16 waves.
// ---------------------------------------------------------------------------
__global__ void fc1_spike_wmma_kernel(const h16* __restrict__ s5,
                                      const h16* __restrict__ Bpack,
                                      const float* __restrict__ bf1,
                                      h16* __restrict__ s6) {
  const int b    = blockIdx.x;        // 0..15
  const int mi   = b & 1;             // batch tile
  const int j    = b >> 1;            // output-channel tile 0..7
  const int lane = threadIdx.x & 31;
  const int hi   = (lane >= 16) ? 1 : 0;
  const int m16  = lane & 15;
  const int nbat = mi * 16 + m16;     // batch row held by this lane's A frag
  const float bs = bf1[j * 16 + m16];

  v8f u;
#pragma unroll
  for (int i = 0; i < 8; ++i) u[i] = 0.f;

  for (int t = 0; t < TSTEPS; ++t) {
    const h16* src = s5 + ((size_t)nbat * TSTEPS + t) * 512;
    v8f c;
#pragma unroll
    for (int i = 0; i < 8; ++i) c[i] = bs;
    for (int kc = 0; kc < 16; ++kc) {
      v16h a;
#pragma unroll
      for (int idx = 0; idx < 16; ++idx) {
        int k = kc * 32 + hi * 8 + idx + ((idx >= 8) ? 8 : 0);
        a[idx] = src[k];
      }
      const v16h* bp = (const v16h*)(Bpack + ((size_t)(kc * 8 + j) * 32 + lane) * 16);
      c = __builtin_amdgcn_wmma_f32_16x16x32_f16(false, a, false, bp[0],
                                                 (short)0, c, false, false);
    }
#pragma unroll
    for (int i = 0; i < 8; ++i) {
      int nn = mi * 16 + i + hi * 8;  // C/D row -> batch index
      u[i] += c[i];
      float s = (u[i] >= 1.f) ? 1.f : 0.f;
      u[i] -= s;
      s6[((size_t)nn * TSTEPS + t) * 128 + j * 16 + m16] = (h16)s;
    }
  }
}

// ---------------------------------------------------------------------------
// fc2 (128->2) + spike + time-mean. 64 neurons total -> one tiny block, fp32.
// ---------------------------------------------------------------------------
__global__ void fc2_out_kernel(const h16* __restrict__ s6,
                               const float* __restrict__ Wf2,
                               const float* __restrict__ bf2,
                               float* __restrict__ out) {
  int tid = threadIdx.x;
  if (tid >= 64) return;
  int n = tid >> 1, o = tid & 1;
  float u = 0.f, acc = 0.f;
  for (int t = 0; t < TSTEPS; ++t) {
    const h16* src = s6 + ((size_t)n * TSTEPS + t) * 128;
    float y = bf2[o];
#pragma unroll 8
    for (int c = 0; c < 128; ++c) y += Wf2[o * 128 + c] * (float)src[c];
    u += y;
    float s = (u >= 1.f) ? 1.f : 0.f;
    u -= s;
    acc += s;
  }
  out[n * 2 + o] = acc / (float)TSTEPS;
}

// ---------------------------------------------------------------------------
extern "C" void kernel_launch(void* const* d_in, const int* in_sizes, int n_in,
                              void* d_out, int out_size, void* d_ws, size_t ws_size,
                              hipStream_t stream) {
  const float* x   = (const float*)d_in[0];
  const float* Wc1 = (const float*)d_in[1];
  const float* bc1 = (const float*)d_in[2];
  const float* Wc2 = (const float*)d_in[3];
  const float* bc2 = (const float*)d_in[4];
  const float* Wc3 = (const float*)d_in[5];
  const float* bc3 = (const float*)d_in[6];
  const float* Wf1 = (const float*)d_in[7];
  const float* bf1 = (const float*)d_in[8];
  const float* Wf2 = (const float*)d_in[9];
  const float* bf2 = (const float*)d_in[10];
  float* out = (float*)d_out;

  char* ws = (char*)d_ws;
  size_t off = 0;
  auto alloc = [&](size_t bytes) {
    char* p = ws + off;
    off = (off + bytes + 255) & ~(size_t)255;
    return p;
  };
  const size_t NT = (size_t)NB * TSTEPS;
  h16* s1   = (h16*)alloc(NT * 16 * 784 * sizeof(h16));   // [N][T][16][28][28]
  h16* s2   = (h16*)alloc(NT * 32 * 576 * sizeof(h16));   // [N][T][32][24][24]
  h16* s3   = (h16*)alloc(NT * 32 * 144 * sizeof(h16));   // [N][T][32][12][12]
  h16* s4   = (h16*)alloc(NT * 32 *  64 * sizeof(h16));   // [N][T][32][8][8]
  h16* s5   = (h16*)alloc(NT * 512       * sizeof(h16));  // [N][T][512]
  h16* s6   = (h16*)alloc(NT * 128       * sizeof(h16));  // [N][T][128]
  h16* W2p  = (h16*)alloc((size_t)13 * 2 * 512 * sizeof(h16));
  h16* W3p  = (h16*)alloc((size_t)25 * 2 * 512 * sizeof(h16));
  h16* W1fp = (h16*)alloc((size_t)16 * 8 * 512 * sizeof(h16));
  (void)ws_size; (void)in_sizes; (void)n_in; (void)out_size;

  // Weight prepacks (f32 -> WMMA-fragment-ordered f16)
  prepack_kernel<<<(13 * 2 * 512 + 255) / 256, 256, 0, stream>>>(Wc2, W2p, 13, 2, 16, 400);
  prepack_kernel<<<(25 * 2 * 512 + 255) / 256, 256, 0, stream>>>(Wc3, W3p, 25, 2, 32, 800);
  prepack_kernel<<<(16 * 8 * 512 + 255) / 256, 256, 0, stream>>>(Wf1, W1fp, 16, 8, 0, 512);

  // Layer pipeline (all on one stream -> serialized)
  conv1_spike_kernel<<<NB * 16, 256, 0, stream>>>(x, Wc1, bc1, s1);
  conv_spike_wmma_kernel<16, 28, 28, 24, 24, 13>
      <<<dim3(9, NB), 128, 0, stream>>>(s1, W2p, bc2, s2);     // 36 patches/n
  pool_spike_kernel<24, 24>
      <<<(NB * 32 * 144 + 255) / 256, 256, 0, stream>>>(s2, s3);
  conv_spike_wmma_kernel<32, 12, 12, 8, 8, 25>
      <<<dim3(1, NB), 128, 0, stream>>>(s3, W3p, bc3, s4);     // 4 patches/n
  pool_spike_kernel<8, 8>
      <<<(NB * 32 * 16 + 255) / 256, 256, 0, stream>>>(s4, s5);
  fc1_spike_wmma_kernel<<<16, 32, 0, stream>>>(s5, W1fp, bf1, s6);
  fc2_out_kernel<<<1, 64, 0, stream>>>(s6, Wf2, bf2, out);
}
